// Model_56392920596884
// MI455X (gfx1250) — matrix operand order repaired and checked by reading, not compiled, not run on hardware
//
#include <hip/hip_runtime.h>
#include <stdint.h>
#include <stddef.h>

// ---------------- model dims ----------------
#define Hh 64
#define Bb 512
#define HB (Hh*Bb)
#define NP 22
#define LSTM 256

typedef __attribute__((ext_vector_type(16))) __bf16 v16bf;
typedef __attribute__((ext_vector_type(8)))  float v8f;
typedef int vint4_ __attribute__((vector_size(16)));

union BF16Tile { uint4 q[2]; v16bf v; };

#if defined(__has_builtin)
#if __has_builtin(__builtin_amdgcn_global_load_async_to_lds_b128) && \
    __has_builtin(__builtin_amdgcn_s_wait_asynccnt)
#define HAVE_ASYNC_LDS 1
#endif
#endif
#ifndef HAVE_ASYNC_LDS
#define HAVE_ASYNC_LDS 0
#endif

#if HAVE_ASYNC_LDS
#define ASYNC_G2L_B128(gp, lp)                                        \
  __builtin_amdgcn_global_load_async_to_lds_b128(                     \
      (__attribute__((address_space(1))) vint4_*)(void*)(gp),         \
      (__attribute__((address_space(3))) vint4_*)(void*)(lp), 0, 0)
#endif

__device__ __forceinline__ float wave_sum(float v){
#pragma unroll
  for (int m = 16; m; m >>= 1) v += __shfl_xor(v, m, 32);
  return v;
}
__device__ __forceinline__ float sigm(float x){ return 1.f / (1.f + expf(-x)); }

// ============================================================
// Simple bf16 WMMA GEMM (one wave per 16x16 tile). Used for the
// latency-bound recurrent GEMM (max wave parallelism) and the
// N=176 head GEMMs. C = A@W^T (+bias[n]) (+init[M,N]).
// ============================================================
__global__ void __launch_bounds__(256)
gemm_bf16_wmma(const __bf16* __restrict__ A, int lda,
               const __bf16* __restrict__ W, int ldw,
               float* __restrict__ C, int ldc,
               const float* __restrict__ bias,
               const float* __restrict__ init, int ldinit,
               int K, int tiles_n, int total_waves)
{
  const int wave = blockIdx.x * 8 + (threadIdx.x >> 5);
  if (wave >= total_waves) return;                 // wave-uniform: EXEC stays all-ones
  const int lane  = threadIdx.x & 31;
  const int l15   = lane & 15;
  const int khalf = lane >> 4;
  const int tn = wave % tiles_n;
  const int tm = wave / tiles_n;
  const int Mb = tm * 16, Nb = tn * 16;
  const int col = Nb + l15;

  v8f acc;
#pragma unroll
  for (int r = 0; r < 8; ++r) {
    const int row = Mb + r + 8 * khalf;
    float a0 = bias ? bias[col] : 0.f;
    if (init) a0 += init[(size_t)row * ldinit + col];
    acc[r] = a0;
  }

  const __bf16* Abase = A + (size_t)(Mb + l15) * lda;
  const __bf16* Bbase = W + (size_t)col * ldw;

  for (int kb = 0; kb < K; kb += 32) {
    BF16Tile a, b;
    // A 16x32 tile: lane m holds 16 contiguous K values starting at kb + 16*khalf
    a.q[0] = *(const uint4*)(Abase + kb + 16 * khalf);
    a.q[1] = *(const uint4*)(Abase + kb + 16 * khalf + 8);
    // B 32x16 tile: lane n holds 16 contiguous K values starting at kb + 16*khalf
    const __bf16* bp = Bbase + kb + 16 * khalf;
    b.q[0] = *(const uint4*)(bp);
    b.q[1] = *(const uint4*)(bp + 8);
    acc = __builtin_amdgcn_wmma_f32_16x16x32_bf16(
        false, a.v, false, b.v, (short)0, acc, false, false);
  }

#pragma unroll
  for (int r = 0; r < 8; ++r) {
    const int row = Mb + r + 8 * khalf;
    C[(size_t)row * ldc + col] = acc[r];
  }
}

// ============================================================
// Tiled bf16 WMMA GEMM for the big parallel GEMMs.
// Workgroup = 8 waves computes a 128x128 tile; BK=32 double-buffered
// LDS staging via async global->LDS (ASYNCcnt) when available.
// Requires M%128==0, N%128==0, K%32==0.
// ============================================================
__global__ void __launch_bounds__(256)
gemm_bf16_tiled(const __bf16* __restrict__ A, int lda,
                const __bf16* __restrict__ W, int ldw,
                float* __restrict__ C, int ldc,
                const float* __restrict__ bias,
                int K, int blocks_n)
{
  __shared__ __align__(16) unsigned char smem[2][16384]; // [buf][A:8KB | B:8KB]

  const int bid = blockIdx.x;
  const int bn = bid % blocks_n, bm = bid / blocks_n;
  const int Mb = bm * 128, Nb = bn * 128;
  const int tid = threadIdx.x;
  const int w = tid >> 5, lane = tid & 31;
  const int l15 = lane & 15, khalf = lane >> 4;

  // staging: 1024 x 16B chunks per k-step (A 512 + B 512); 4 per thread
  const __bf16* gbase[4];
  int loff[4];
#pragma unroll
  for (int i = 0; i < 4; ++i) {
    const int c = tid + 256 * i;
    const bool isA = c < 512;
    const int cc = isA ? c : c - 512;
    const int rowl = cc >> 2, sub = cc & 3;
    gbase[i] = isA ? (A + (size_t)(Mb + rowl) * lda + sub * 8)
                   : (W + (size_t)(Nb + rowl) * ldw + sub * 8);
    loff[i] = (isA ? 0 : 8192) + rowl * 64 + sub * 16;
  }

  v8f acc[8];
  const int col0 = Nb + l15;
#pragma unroll
  for (int tn = 0; tn < 8; ++tn) {
    const float bv = bias ? bias[col0 + tn * 16] : 0.f;
#pragma unroll
    for (int r = 0; r < 8; ++r) acc[tn][r] = bv;
  }

  const int S = K >> 5;
  const int arow = w * 16 + l15;

#if HAVE_ASYNC_LDS
  // prologue: stage k-step 0 into buffer 0
#pragma unroll
  for (int i = 0; i < 4; ++i)
    ASYNC_G2L_B128(gbase[i], &smem[0][0] + loff[i]);
  __builtin_amdgcn_s_wait_asynccnt(0);
  __syncthreads();

  for (int kb = 0; kb < S; ++kb) {
    const int cur = kb & 1;
    if (kb + 1 < S) {
#pragma unroll
      for (int i = 0; i < 4; ++i)
        ASYNC_G2L_B128(gbase[i] + (kb + 1) * 32, &smem[cur ^ 1][0] + loff[i]);
    }
    {
      const unsigned char* sb = &smem[cur][0];
      BF16Tile a;
      a.q[0] = *(const uint4*)(sb + arow * 64 + khalf * 32);
      a.q[1] = *(const uint4*)(sb + arow * 64 + 16 + khalf * 32);
#pragma unroll
      for (int tn = 0; tn < 8; ++tn) {
        BF16Tile b;
        const unsigned char* bp = sb + 8192 + (tn * 16 + l15) * 64 + khalf * 32;
        b.q[0] = *(const uint4*)(bp);
        b.q[1] = *(const uint4*)(bp + 16);
        acc[tn] = __builtin_amdgcn_wmma_f32_16x16x32_bf16(
            false, a.v, false, b.v, (short)0, acc[tn], false, false);
      }
    }
    if (kb + 1 < S) {
      __builtin_amdgcn_s_wait_asynccnt(0);  // next buffer staged (per-wave)
      __syncthreads();                      // ...and visible to all waves; also
    }                                       // guarantees cur buffer free for reuse
  }
#else
  // synchronous fallback: single-buffered stage + barrier
  for (int kb = 0; kb < S; ++kb) {
    uint4 rch[4];
#pragma unroll
    for (int i = 0; i < 4; ++i) rch[i] = *(const uint4*)(gbase[i] + kb * 32);
    __syncthreads();
#pragma unroll
    for (int i = 0; i < 4; ++i) *(uint4*)(&smem[0][0] + loff[i]) = rch[i];
    __syncthreads();
    {
      const unsigned char* sb = &smem[0][0];
      BF16Tile a;
      a.q[0] = *(const uint4*)(sb + arow * 64 + khalf * 32);
      a.q[1] = *(const uint4*)(sb + arow * 64 + 16 + khalf * 32);
#pragma unroll
      for (int tn = 0; tn < 8; ++tn) {
        BF16Tile b;
        const unsigned char* bp = sb + 8192 + (tn * 16 + l15) * 64 + khalf * 32;
        b.q[0] = *(const uint4*)(bp);
        b.q[1] = *(const uint4*)(bp + 16);
        acc[tn] = __builtin_amdgcn_wmma_f32_16x16x32_bf16(
            false, a.v, false, b.v, (short)0, acc[tn], false, false);
      }
    }
    __syncthreads();
  }
#endif

#pragma unroll
  for (int tn = 0; tn < 8; ++tn) {
    const int col = Nb + tn * 16 + l15;
#pragma unroll
    for (int r = 0; r < 8; ++r) {
      const int row = Mb + w * 16 + r + 8 * khalf;
      C[(size_t)row * ldc + col] = acc[tn][r];
    }
  }
}

// ============================================================
// Conversions / padding
// ============================================================
__global__ void f32_to_bf16_kernel(const float* __restrict__ x, __bf16* __restrict__ y, int n)
{
  int i = blockIdx.x * blockDim.x + threadIdx.x;
  if (i < n) y[i] = (__bf16)x[i];
}

__global__ void f32_to_bf16_padrows_kernel(const float* __restrict__ x, __bf16* __restrict__ y,
                                           int rows, int cols, int rows_pad)
{
  int i = blockIdx.x * blockDim.x + threadIdx.x;
  if (i < rows_pad * cols) {
    int r = i / cols;
    y[i] = (r < rows) ? (__bf16)x[i] : (__bf16)0.f;
  }
}

__global__ void pad_vec_kernel(const float* __restrict__ x, float* __restrict__ y, int n, int npad)
{
  int i = blockIdx.x * blockDim.x + threadIdx.x;
  if (i < npad) y[i] = (i < n) ? x[i] : 0.f;
}

// ============================================================
// Embed + attention front-end: one wave per (h,b) item.
// ============================================================
#define K1W 4
__global__ void __launch_bounds__(128)
embed_attn_kernel(const float* __restrict__ ms, const float* __restrict__ ps,
                  const float* __restrict__ bsit, const float* __restrict__ pst,
                  const float* __restrict__ ats,
                  const float* __restrict__ mW, const float* __restrict__ mb,
                  const float* __restrict__ pW, const float* __restrict__ pb,
                  const float* __restrict__ bW, const float* __restrict__ bb,
                  const float* __restrict__ tW, const float* __restrict__ tb,
                  const float* __restrict__ wsW, const float* __restrict__ wsb,
                  const float* __restrict__ asW, const float* __restrict__ asb,
                  const float* __restrict__ g_m, const float* __restrict__ b_m,
                  const float* __restrict__ g_p, const float* __restrict__ b_p,
                  const float* __restrict__ g_b, const float* __restrict__ b_b,
                  __bf16* __restrict__ cat_bf, float* __restrict__ idx_out)
{
  __shared__ float s_t[K1W][NP][48];
  __shared__ float s_aws[K1W][NP][48];
  __shared__ float s_pe[K1W][48];
  __shared__ float s_pws[K1W][48];
  __shared__ float s_att[K1W][NP];
  __shared__ int   s_idx[K1W][3];

  const int w    = threadIdx.x >> 5;
  const int lane = threadIdx.x & 31;
  const int item = blockIdx.x * K1W + w;
  __bf16* crow = cat_bf + (size_t)item * 384;
  const bool lo = (lane < 16);

  // ---- three 64-wide LN+ReLU embeddings -> cat cols [0,192) ----
  {
    const float* xs[3]  = { ms + (size_t)item*24, ps + (size_t)item*29, bsit + (size_t)item*18 };
    const float* Ws[3]  = { mW, pW, bW };
    const float* bv[3]  = { mb, pb, bb };
    const float* gs[3]  = { g_m, g_p, g_b };
    const float* bl[3]  = { b_m, b_p, b_b };
    const int    Ds[3]  = { 24, 29, 18 };
    for (int e = 0; e < 3; ++e) {
      const float* x = xs[e]; const float* W = Ws[e]; const int D = Ds[e];
      float y0 = bv[e][lane], y1 = bv[e][lane + 32];
      for (int k = 0; k < D; ++k) {
        float xv = x[k];
        y0 += xv * W[lane * D + k];
        y1 += xv * W[(lane + 32) * D + k];
      }
      float mean = wave_sum(y0 + y1) * (1.f / 64.f);
      float d0 = y0 - mean, d1 = y1 - mean;
      float var = wave_sum(d0 * d0 + d1 * d1) * (1.f / 64.f);
      float inv = rsqrtf(var + 1e-5f);
      float r0 = fmaxf(d0 * inv * gs[e][lane]      + bl[e][lane],      0.f);
      float r1 = fmaxf(d1 * inv * gs[e][lane + 32] + bl[e][lane + 32], 0.f);
      crow[e * 64 + lane]      = (__bf16)r0;
      crow[e * 64 + lane + 32] = (__bf16)r1;
    }
  }

  // ---- player embedding (48, ReLU) ----
  {
    const float* x = pst + (size_t)item * 29;
    float e0 = tb[lane];
    float e1 = lo ? tb[lane + 32] : 0.f;
    for (int k = 0; k < 29; ++k) {
      float xv = x[k];
      e0 += xv * tW[lane * 29 + k];
      if (lo) e1 += xv * tW[(lane + 32) * 29 + k];
    }
    s_pe[w][lane] = fmaxf(e0, 0.f);
    if (lo) s_pe[w][lane + 32] = fmaxf(e1, 0.f);
  }

  // ---- team embeddings (22 x 48, ReLU) ----
  for (int n = 0; n < NP; ++n) {
    const float* x = ats + ((size_t)item * NP + n) * 29;
    float e0 = tb[lane];
    float e1 = lo ? tb[lane + 32] : 0.f;
    for (int k = 0; k < 29; ++k) {
      float xv = x[k];
      e0 += xv * tW[lane * 29 + k];
      if (lo) e1 += xv * tW[(lane + 32) * 29 + k];
    }
    s_t[w][n][lane] = fmaxf(e0, 0.f);
    if (lo) s_t[w][n][lane + 32] = fmaxf(e1, 0.f);
  }
  __syncthreads();

  // ---- ws projections ----
  for (int n = 0; n < NP; ++n) {
    float a0 = wsb[lane];
    float a1 = lo ? wsb[lane + 32] : 0.f;
    for (int k = 0; k < 48; ++k) {
      float tv = s_t[w][n][k];
      a0 += tv * wsW[lane * 48 + k];
      if (lo) a1 += tv * wsW[(lane + 32) * 48 + k];
    }
    s_aws[w][n][lane] = a0;
    if (lo) s_aws[w][n][lane + 32] = a1;
  }
  {
    float q0 = wsb[lane];
    float q1 = lo ? wsb[lane + 32] : 0.f;
    for (int k = 0; k < 48; ++k) {
      float pv = s_pe[w][k];
      q0 += pv * wsW[lane * 48 + k];
      if (lo) q1 += pv * wsW[(lane + 32) * 48 + k];
    }
    s_pws[w][lane] = q0;
    if (lo) s_pws[w][lane + 32] = q1;
  }
  __syncthreads();

  // ---- attention scores (leaky relu 0.01) ----
  if (lane < NP) {
    float a = asb[0];
    for (int k = 0; k < 48; ++k)
      a += asW[k] * s_pws[w][k] + asW[48 + k] * s_aws[w][lane][k];
    s_att[w][lane] = (a > 0.f) ? a : 0.01f * a;
  }
  __syncthreads();

  // ---- softmax, weighted elu sum, 3-smallest indices ----
  float mx = -1e30f;
  for (int n = 0; n < NP; ++n) mx = fmaxf(mx, s_att[w][n]);
  float den = 0.f;
  for (int n = 0; n < NP; ++n) den += expf(s_att[w][n] - mx);
  {
    float v0 = 0.f, v1 = 0.f;
    for (int n = 0; n < NP; ++n) {
      float p = expf(s_att[w][n] - mx);
      v0 += p * s_aws[w][n][lane];
      if (lo) v1 += p * s_aws[w][n][lane + 32];
    }
    v0 /= den; v1 /= den;
    v0 = (v0 > 0.f) ? v0 : (expf(v0) - 1.f);
    v1 = (v1 > 0.f) ? v1 : (expf(v1) - 1.f);
    crow[192 + lane] = (__bf16)v0;
    if (lo) crow[192 + lane + 32] = (__bf16)v1;
  }
  if (lane < NP) {
    float a = s_att[w][lane];
    int rank = 0;
    for (int m = 0; m < NP; ++m) {
      float am = s_att[w][m];
      rank += (am < a) || (am == a && m < lane);
    }
    if (rank < 3) s_idx[w][rank] = lane;
  }
  __syncthreads();
#pragma unroll
  for (int k = 0; k < 3; ++k) {
    int n = s_idx[w][k];
    crow[240 + 48 * k + lane] = (__bf16)s_t[w][n][lane];
    if (lo) crow[240 + 48 * k + lane + 32] = (__bf16)s_t[w][n][lane + 32];
  }
  if (lane < 3) idx_out[(size_t)item * 3 + lane] = (float)s_idx[w][lane];
}

// ============================================================
// LayerNorm(+ReLU) over Dv dims of a [R, ld] matrix, one wave per row.
// ============================================================
__global__ void __launch_bounds__(256)
ln_relu_kernel(const float* __restrict__ X, int ld, int D, int Dv,
               const float* __restrict__ g, const float* __restrict__ b,
               float* __restrict__ out_f32, __bf16* __restrict__ out_bf, int R)
{
  const int row = blockIdx.x * 8 + (threadIdx.x >> 5);
  if (row >= R) return;
  const int lane = threadIdx.x & 31;
  const float* x = X + (size_t)row * ld;
  float s = 0.f, ss = 0.f;
  for (int d = lane; d < Dv; d += 32) { float v = x[d]; s += v; ss += v * v; }
  s = wave_sum(s); ss = wave_sum(ss);
  const float mean = s / (float)Dv;
  const float var  = ss / (float)Dv - mean * mean;
  const float inv  = rsqrtf(var + 1e-5f);
  for (int d = lane; d < D; d += 32) {
    float y = 0.f;
    if (d < Dv) y = fmaxf((x[d] - mean) * inv * g[d] + b[d], 0.f);
    if (out_f32) out_f32[(size_t)row * ld + d] = y;
    if (out_bf)  out_bf[(size_t)row * ld + d]  = (__bf16)y;
  }
}

// ============================================================
// LSTM cell elementwise update
// ============================================================
__global__ void __launch_bounds__(256)
lstm_cell_kernel(const float* __restrict__ z, float* __restrict__ c,
                 float* __restrict__ h_f32, __bf16* __restrict__ h_bf)
{
  const int i = blockIdx.x * blockDim.x + threadIdx.x;
  if (i >= Bb * LSTM) return;
  const int b = i >> 8, u = i & 255;
  const float* zb = z + (size_t)b * 1024;
  const float zi = zb[u], zf = zb[256 + u], zg = zb[512 + u], zo = zb[768 + u];
  const float cc = sigm(zf) * c[i] + sigm(zi) * tanhf(zg);
  const float hh = sigm(zo) * tanhf(cc);
  c[i] = cc;
  h_f32[i] = hh;
  h_bf[i] = (__bf16)hh;
}

// ============================================================
// Tiny second-layer heads (VALU)
// ============================================================
__global__ void __launch_bounds__(256)
head_policy_kernel(const float* __restrict__ X, int ldx,
                   const float* __restrict__ W2, const float* __restrict__ b2,
                   const float* __restrict__ avail, int nout,
                   float* __restrict__ out)
{
  const int row = blockIdx.x * blockDim.x + threadIdx.x;
  if (row >= HB) return;
  const float* x = X + (size_t)row * ldx;
  float lg[12];
  float mx = -1e30f;
  for (int j = 0; j < nout; ++j) {
    float s = b2[j];
    for (int k = 0; k < 164; ++k) s += x[k] * W2[j * 164 + k];
    if (avail) s += (avail[(size_t)row * nout + j] - 1.f) * 1e7f;
    lg[j] = s;
    mx = fmaxf(mx, s);
  }
  float den = 0.f;
  for (int j = 0; j < nout; ++j) { lg[j] = expf(lg[j] - mx); den += lg[j]; }
  for (int j = 0; j < nout; ++j) out[(size_t)row * nout + j] = lg[j] / den;
}

__global__ void __launch_bounds__(256)
head_value_kernel(const float* __restrict__ X, int ldx,
                  const float* __restrict__ W2, float* __restrict__ out)
{
  const int row = blockIdx.x * blockDim.x + threadIdx.x;
  if (row >= HB) return;
  const float* x = X + (size_t)row * ldx;
  float s = 0.f;
  for (int k = 0; k < 164; ++k) s += x[k] * W2[k];
  out[row] = s;
}

// ============================================================
// Host side
// ============================================================
static inline void launch_gemm(hipStream_t s,
                               const __bf16* A, int lda, const __bf16* W, int ldw,
                               float* C, int ldc, const float* bias,
                               const float* init, int ldinit,
                               int M, int N, int K)
{
  const int tiles_m = M / 16, tiles_n = N / 16;
  const int waves = tiles_m * tiles_n;
  const int blocks = (waves + 7) / 8;
  gemm_bf16_wmma<<<blocks, 256, 0, s>>>(A, lda, W, ldw, C, ldc, bias, init, ldinit,
                                        K, tiles_n, waves);
}

static inline void launch_gemm_tiled(hipStream_t s,
                                     const __bf16* A, int lda, const __bf16* W, int ldw,
                                     float* C, int ldc, const float* bias,
                                     int M, int N, int K)
{
  const int bm = M / 128, bn = N / 128;
  gemm_bf16_tiled<<<bm * bn, 256, 0, s>>>(A, lda, W, ldw, C, ldc, bias, K, bn);
}

extern "C" void kernel_launch(void* const* d_in, const int* in_sizes, int n_in,
                              void* d_out, int out_size, void* d_ws, size_t ws_size,
                              hipStream_t stream)
{
  (void)in_sizes; (void)n_in; (void)out_size; (void)ws_size;
  const float* ms    = (const float*)d_in[0];   // [H,B,24]
  const float* psit  = (const float*)d_in[1];   // [H,B,29]
  const float* bsit  = (const float*)d_in[2];   // [H,B,18]
  const float* pst   = (const float*)d_in[3];   // [H,B,29]
  const float* ats   = (const float*)d_in[4];   // [H,B,22,29]
  const float* avail = (const float*)d_in[5];   // [H,B,12]
  const float* h0    = (const float*)d_in[6];   // [B,256]
  const float* c0    = (const float*)d_in[7];   // [B,256]
  const float* mW  = (const float*)d_in[8];  const float* mb  = (const float*)d_in[9];
  const float* pW  = (const float*)d_in[10]; const float* pb  = (const float*)d_in[11];
  const float* bW  = (const float*)d_in[12]; const float* bb  = (const float*)d_in[13];
  const float* tW  = (const float*)d_in[14]; const float* tb  = (const float*)d_in[15];
  const float* wsW = (const float*)d_in[16]; const float* wsb = (const float*)d_in[17];
  const float* asW = (const float*)d_in[18]; const float* asb = (const float*)d_in[19];
  const float* catW= (const float*)d_in[20]; const float* catb= (const float*)d_in[21];
  const float* Wih = (const float*)d_in[22]; const float* Whh = (const float*)d_in[23];
  const float* bih = (const float*)d_in[24]; const float* bhh = (const float*)d_in[25];
  const float* a1W = (const float*)d_in[26]; const float* a1b = (const float*)d_in[27];
  const float* a2W = (const float*)d_in[28]; const float* a2b = (const float*)d_in[29];
  const float* m1W = (const float*)d_in[30]; const float* m1b = (const float*)d_in[31];
  const float* m2W = (const float*)d_in[32]; const float* m2b = (const float*)d_in[33];
  const float* v1W = (const float*)d_in[34]; const float* v1b = (const float*)d_in[35];
  const float* v2W = (const float*)d_in[36];
  const float* lnm_g = (const float*)d_in[37]; const float* lnm_b = (const float*)d_in[38];
  const float* lnp_g = (const float*)d_in[39]; const float* lnp_b = (const float*)d_in[40];
  const float* lnb_g = (const float*)d_in[41]; const float* lnb_b = (const float*)d_in[42];
  const float* lncat_g = (const float*)d_in[43]; const float* lncat_b = (const float*)d_in[44];
  const float* lna_g = (const float*)d_in[45]; const float* lna_b = (const float*)d_in[46];
  const float* lnpm_g = (const float*)d_in[47]; const float* lnpm_b = (const float*)d_in[48];
  const float* lnv_g = (const float*)d_in[49]; const float* lnv_b = (const float*)d_in[50];

  float* out = (float*)d_out;
  float* out_prob  = out;                         // [H,B,12]
  float* out_probm = out + (size_t)HB * 12;       // [H,B,8]
  float* out_v     = out_probm + (size_t)HB * 8;  // [H,B,1]
  float* out_hT    = out_v + HB;                  // [B,256]
  float* out_cT    = out_hT + (size_t)Bb * LSTM;  // [B,256]
  float* out_idx   = out_cT + (size_t)Bb * LSTM;  // [HB,3] (as float)

  char* ws = (char*)d_ws;
  size_t off = 0;
  auto alloc = [&](size_t bytes) -> char* {
    char* p = ws + off;
    off = (off + bytes + 255) & ~(size_t)255;
    return p;
  };
  __bf16* catW_bf = (__bf16*)alloc((size_t)256 * 384 * 2);
  __bf16* Wih_bf  = (__bf16*)alloc((size_t)1024 * 256 * 2);
  __bf16* Whh_bf  = (__bf16*)alloc((size_t)1024 * 256 * 2);
  __bf16* a1W_bf  = (__bf16*)alloc((size_t)176 * 256 * 2);
  __bf16* m1W_bf  = (__bf16*)alloc((size_t)176 * 256 * 2);
  __bf16* v1W_bf  = (__bf16*)alloc((size_t)176 * 256 * 2);
  float*  a1b_p   = (float*)alloc(176 * 4);
  float*  m1b_p   = (float*)alloc(176 * 4);
  float*  v1b_p   = (float*)alloc(176 * 4);
  __bf16* cat_bf  = (__bf16*)alloc((size_t)HB * 384 * 2);
  float*  tmp1    = (float*)alloc((size_t)HB * 256 * 4);
  __bf16* catn_bf = (__bf16*)alloc((size_t)HB * 256 * 2);
  float*  Xpre    = (float*)alloc((size_t)HB * 1024 * 4);
  __bf16* h_bf    = (__bf16*)alloc((size_t)Bb * LSTM * 2);
  float*  c_st    = (float*)alloc((size_t)Bb * LSTM * 4);
  float*  zbuf    = (float*)alloc((size_t)Bb * 1024 * 4);
  float*  outseq  = (float*)alloc((size_t)HB * LSTM * 4);
  __bf16* out_bf  = (__bf16*)alloc((size_t)HB * LSTM * 2);
  float*  hpre    = (float*)alloc((size_t)HB * 176 * 4);
  float*  hln     = (float*)alloc((size_t)HB * 176 * 4);

  auto cvt = [&](const float* src, __bf16* dst, int n) {
    f32_to_bf16_kernel<<<(n + 255) / 256, 256, 0, stream>>>(src, dst, n);
  };
  cvt(catW, catW_bf, 256 * 384);
  cvt(Wih,  Wih_bf,  1024 * 256);
  cvt(Whh,  Whh_bf,  1024 * 256);
  f32_to_bf16_padrows_kernel<<<(176*256 + 255)/256, 256, 0, stream>>>(a1W, a1W_bf, 164, 256, 176);
  f32_to_bf16_padrows_kernel<<<(176*256 + 255)/256, 256, 0, stream>>>(m1W, m1W_bf, 164, 256, 176);
  f32_to_bf16_padrows_kernel<<<(176*256 + 255)/256, 256, 0, stream>>>(v1W, v1W_bf, 164, 256, 176);
  pad_vec_kernel<<<1, 256, 0, stream>>>(a1b, a1b_p, 164, 176);
  pad_vec_kernel<<<1, 256, 0, stream>>>(m1b, m1b_p, 164, 176);
  pad_vec_kernel<<<1, 256, 0, stream>>>(v1b, v1b_p, 164, 176);
  cvt(h0, h_bf, Bb * LSTM);
  (void)hipMemcpyAsync(c_st, c0, (size_t)Bb * LSTM * 4, hipMemcpyDeviceToDevice, stream);

  // ---- front-end ----
  embed_attn_kernel<<<HB / K1W, 32 * K1W, 0, stream>>>(
      ms, psit, bsit, pst, ats,
      mW, mb, pW, pb, bW, bb, tW, tb, wsW, wsb, asW, asb,
      lnm_g, lnm_b, lnp_g, lnp_b, lnb_g, lnb_b,
      cat_bf, out_idx);

  // ---- cat projection (tiled async WMMA) + LN+ReLU ----
  launch_gemm_tiled(stream, cat_bf, 384, catW_bf, 384, tmp1, 256, catb, HB, 256, 384);
  ln_relu_kernel<<<HB / 8, 256, 0, stream>>>(tmp1, 256, 256, 256, lncat_g, lncat_b,
                                             nullptr, catn_bf, HB);

  // ---- hoisted input-gate GEMM for all timesteps (tiled async WMMA) ----
  launch_gemm_tiled(stream, catn_bf, 256, Wih_bf, 256, Xpre, 1024, bih, HB, 1024, 256);

  // ---- sequential LSTM ----
  for (int t = 0; t < Hh; ++t) {
    launch_gemm(stream, h_bf, 256, Whh_bf, 256, zbuf, 1024, bhh,
                Xpre + (size_t)t * Bb * 1024, 1024, Bb, 1024, 256);
    lstm_cell_kernel<<<(Bb * LSTM) / 256, 256, 0, stream>>>(
        zbuf, c_st, outseq + (size_t)t * Bb * LSTM, h_bf);
  }
  (void)hipMemcpyAsync(out_hT, outseq + (size_t)(Hh - 1) * Bb * LSTM,
                       (size_t)Bb * LSTM * 4, hipMemcpyDeviceToDevice, stream);
  (void)hipMemcpyAsync(out_cT, c_st, (size_t)Bb * LSTM * 4, hipMemcpyDeviceToDevice, stream);

  // ---- heads ----
  cvt(outseq, out_bf, HB * LSTM);
  launch_gemm(stream, out_bf, 256, a1W_bf, 256, hpre, 176, a1b_p, nullptr, 0, HB, 176, 256);
  ln_relu_kernel<<<HB / 8, 256, 0, stream>>>(hpre, 176, 176, 164, lna_g, lna_b, hln, nullptr, HB);
  head_policy_kernel<<<HB / 256, 256, 0, stream>>>(hln, 176, a2W, a2b, avail, 12, out_prob);

  launch_gemm(stream, out_bf, 256, m1W_bf, 256, hpre, 176, m1b_p, nullptr, 0, HB, 176, 256);
  ln_relu_kernel<<<HB / 8, 256, 0, stream>>>(hpre, 176, 176, 164, lnpm_g, lnpm_b, hln, nullptr, HB);
  head_policy_kernel<<<HB / 256, 256, 0, stream>>>(hln, 176, m2W, m2b, nullptr, 8, out_probm);

  launch_gemm(stream, out_bf, 256, v1W_bf, 256, hpre, 176, v1b_p, nullptr, 0, HB, 176, 256);
  ln_relu_kernel<<<HB / 8, 256, 0, stream>>>(hpre, 176, 176, 164, lnv_g, lnv_b, hln, nullptr, HB);
  head_value_kernel<<<HB / 256, 256, 0, stream>>>(hln, 176, v2W, out_v);
}
